// autoencoder_38070590112103
// MI455X (gfx1250) — matrix operand order repaired and checked by reading, not compiled, not run on hardware
//
#include <hip/hip_runtime.h>

// GCN autoencoder for MI455X (gfx1250, wave32, WMMA).
//
// Memory-bound (~2 GB HBM traffic vs ~27 GFLOP of GEMM): all dense GEMMs run
// on v_wmma_f32_16x16x32_f16 (f16 inputs, f32 accumulate), the 192 MB L2
// absorbs the 16x A re-read of the wide decode GEMM, and the edge scatter is
// pure coalesced gather + coalesced f32 atomics with wave-uniform edge
// metadata scalarized onto the SMEM path. Both A and B operands of the GEMM
// are software-pipelined one K-step ahead so no WMMA waits on a just-issued
// load.

typedef __attribute__((ext_vector_type(16))) _Float16 v16h;
typedef __attribute__((ext_vector_type(8)))  float    v8f;

#define TPB 256

// ---------------------------------------------------------------- utilities
__global__ __launch_bounds__(TPB) void k_fill(float* __restrict__ p, long long n, float v) {
  long long i = (long long)blockIdx.x * TPB + threadIdx.x;
  if (i < n) p[i] = v;
}

__global__ __launch_bounds__(TPB) void k_degcount(const int* __restrict__ dst,
                                                  float* __restrict__ deg, int nE) {
  int e = blockIdx.x * TPB + threadIdx.x;
  if (e < nE) atomicAdd(deg + dst[e], 1.0f);
}

__global__ __launch_bounds__(TPB) void k_rsqrt(float* __restrict__ p, int n) {
  int i = blockIdx.x * TPB + threadIdx.x;
  if (i < n) p[i] = rsqrtf(p[i]);   // deg >= 1 always (self loop)
}

// Pre-pack W[K x Nc] (f32 row-major) into the B-fragment lane layout for
// v_wmma_f32_16x16x32_f16 (B 32x16: lane l -> col (l&15), K=(e>>3)*16+(l>>4)*8+(e&7)),
// so GEMM B-loads are two coalesced global_load_b128 per fragment.
__global__ __launch_bounds__(TPB) void k_prepack(const float* __restrict__ W,
                                                 _Float16* __restrict__ Wpk,
                                                 int K, int Nc) {
  int idx = blockIdx.x * TPB + threadIdx.x;
  int total = (Nc >> 4) * (K >> 5) * 512;
  if (idx >= total) return;
  int e    = idx & 15;
  int lane = (idx >> 4) & 31;
  int fi   = idx >> 9;
  int KB   = K >> 5;
  int kb   = fi % KB;
  int cb   = fi / KB;
  int col  = (cb << 4) + (lane & 15);
  int k    = (kb << 5) + ((e >> 3) << 4) + ((lane >> 4) << 3) + (e & 7);
  Wpk[idx] = (_Float16)W[(size_t)k * Nc + col];
}

// ---------------------------------------------------------------- WMMA GEMM
// Convert resident A tile to f16 (ISA 16-bit A 16x32 layout: elems 0..7 ->
// K=kp*8+e, elems 8..15 -> 16+kp*8+e) and issue NJT back-to-back WMMAs
// against already-resident B fragments. Fully branch-free.
template <int NJT>
__device__ __forceinline__ void wmma_step(const float4& a0, const float4& a1,
                                          const float4& a2, const float4& a3,
                                          const v16h (&b)[NJT], v8f (&acc)[NJT]) {
  union { v16h v; _Float16 h[16]; } af;
  af.h[0]=(_Float16)a0.x;  af.h[1]=(_Float16)a0.y;  af.h[2]=(_Float16)a0.z;  af.h[3]=(_Float16)a0.w;
  af.h[4]=(_Float16)a1.x;  af.h[5]=(_Float16)a1.y;  af.h[6]=(_Float16)a1.z;  af.h[7]=(_Float16)a1.w;
  af.h[8]=(_Float16)a2.x;  af.h[9]=(_Float16)a2.y;  af.h[10]=(_Float16)a2.z; af.h[11]=(_Float16)a2.w;
  af.h[12]=(_Float16)a3.x; af.h[13]=(_Float16)a3.y; af.h[14]=(_Float16)a3.z; af.h[15]=(_Float16)a3.w;
#pragma unroll
  for (int j = 0; j < NJT; ++j)
    acc[j] = __builtin_amdgcn_wmma_f32_16x16x32_f16(false, af.v, false, b[j],
                                                    (short)0, acc[j], false, false);
}

// C[M x Nc] = A[M x K] * W (pre-packed f16). 8 waves/block, wave owns a
// 16 x (NJT*16) output strip. Both A tile and B fragments are double-buffered
// one K-step ahead: the loads a WMMA depends on were issued a full iteration
// earlier, so s_wait_loadcnt never gates on a just-issued load.
// FUSE: bit0 = add bias, bit1 = relu.
template <int NJT, int FUSE>
__global__ __launch_bounds__(TPB) void k_gemm(const float* __restrict__ A,
                                              const _Float16* __restrict__ Wpk,
                                              float* __restrict__ Cmat,
                                              const float* __restrict__ bias,
                                              int M, int K, int Nc) {
  const int lane = threadIdx.x & 31;
  const int wave = threadIdx.x >> 5;
  const int kp   = lane >> 4;
  int row = blockIdx.x * 128 + wave * 16 + (lane & 15);
  if (row >= M) row = M - 1;              // branchless clamp: EXEC all-1 for WMMA
  const float* abase = A + (size_t)row * K + kp * 8;

  const int KB = K >> 5;
  const v16h* Bp = (const v16h*)Wpk + (size_t)(blockIdx.y << 2) * KB * 32 + lane;

  v8f acc[NJT];
#pragma unroll
  for (int j = 0; j < NJT; ++j) acc[j] = v8f{};

  // prologue: A tile + B fragments for kb = 0 (lanes l, l+16 together cover
  // one full 128B cacheline of row r per step -> fully-used HBM lines)
  float4 c0 = *(const float4*)(abase);
  float4 c1 = *(const float4*)(abase + 4);
  float4 c2 = *(const float4*)(abase + 16);
  float4 c3 = *(const float4*)(abase + 20);
  v16h bc[NJT];
#pragma unroll
  for (int j = 0; j < NJT; ++j) bc[j] = Bp[(size_t)(j * KB) * 32];

#pragma unroll 2
  for (int kb = 0; kb < KB - 1; ++kb) {
    // issue next step's loads before consuming this step's registers
    const float* ap = abase + (kb + 1) * 32;
    float4 n0 = *(const float4*)(ap);
    float4 n1 = *(const float4*)(ap + 4);
    float4 n2 = *(const float4*)(ap + 16);
    float4 n3 = *(const float4*)(ap + 20);
    v16h bn[NJT];
#pragma unroll
    for (int j = 0; j < NJT; ++j) bn[j] = Bp[(size_t)(j * KB + kb + 1) * 32];
    if (kb + 2 < KB)
      __builtin_prefetch((const void*)(abase + (kb + 2) * 32), 0, 3); // global_prefetch_b8

    wmma_step<NJT>(c0, c1, c2, c3, bc, acc);

    c0 = n0; c1 = n1; c2 = n2; c3 = n3;
#pragma unroll
    for (int j = 0; j < NJT; ++j) bc[j] = bn[j];
  }
  wmma_step<NJT>(c0, c1, c2, c3, bc, acc);   // peeled tail

  // Epilogue: C/D layout lane l -> col (l&15), VGPR v -> row v + (l>>4)*8.
  const int colb = (blockIdx.y << 6) + (lane & 15);
  const int rb   = blockIdx.x * 128 + wave * 16 + kp * 8;
#pragma unroll
  for (int j = 0; j < NJT; ++j) {
    int col = colb + (j << 4);
    float bv = (FUSE & 1) ? bias[col] : 0.0f;
#pragma unroll
    for (int v = 0; v < 8; ++v) {
      int r = rb + v;
      if (r < M) {
        float val = acc[j][v] + bv;
        if (FUSE & 2) val = fmaxf(val, 0.0f);
        Cmat[(size_t)r * Nc + col] = val;
      }
    }
  }
}

// ------------------------------------------------------- edge scatter (GCN)
// TPB is a multiple of C (32/64), so each wave handles exactly one
// (edge, 32-channel chunk): edge id and norm coefficient are wave-uniform.
// readfirstlane pushes src/dst/dinv loads onto the scalar (SMEM) path; the
// vector side is only the coalesced gather + coalesced global_atomic_add_f32.
__global__ __launch_bounds__(TPB) void k_scatter(const float* __restrict__ h,
                                                 const int* __restrict__ src,
                                                 const int* __restrict__ dst,
                                                 const float* __restrict__ dinv,
                                                 float* __restrict__ acc,
                                                 int nE, int C, int logC) {
  long long t = (long long)blockIdx.x * TPB + threadIdx.x;
  long long e = t >> logC;
  if (e >= nE) return;
  int ew = __builtin_amdgcn_readfirstlane((int)e);   // wave-uniform by construction
  int c  = (int)(t & (long long)(C - 1));
  int s  = src[ew];
  int d  = dst[ew];
  float w = dinv[s] * dinv[d];
  atomicAdd(acc + (size_t)d * C + c, h[(size_t)s * C + c] * w);
}

// finalize: out = acc + self_loop(pre * dinv^2) + bias, optional relu
__global__ __launch_bounds__(TPB) void k_finalize(const float* __restrict__ pre,
                                                  const float* __restrict__ acc,
                                                  const float* __restrict__ dinv,
                                                  const float* __restrict__ bias,
                                                  float* __restrict__ out,
                                                  int nN, int C, int logC, int relu) {
  long long idx = (long long)blockIdx.x * TPB + threadIdx.x;
  if (idx >= (long long)nN * C) return;
  int i = (int)(idx >> logC);
  int c = (int)(idx & (long long)(C - 1));
  float di = dinv[i];
  float v = acc[idx] + pre[idx] * di * di + bias[c];
  if (relu) v = fmaxf(v, 0.0f);
  out[idx] = v;
}

// ---------------------------------------------------------------- launcher
extern "C" void kernel_launch(void* const* d_in, const int* in_sizes, int n_in,
                              void* d_out, int out_size, void* d_ws, size_t ws_size,
                              hipStream_t stream) {
  const float* x   = (const float*)d_in[0];
  const int*   ei  = (const int*)d_in[1];
  const float* Wg1 = (const float*)d_in[2]; const float* bg1 = (const float*)d_in[3];
  const float* Wg2 = (const float*)d_in[4]; const float* bg2 = (const float*)d_in[5];
  const float* Wf1 = (const float*)d_in[6]; const float* bf1 = (const float*)d_in[7];
  const float* Wf2 = (const float*)d_in[8]; const float* bf2 = (const float*)d_in[9];

  const int IN_CH = 1024, Z = 32, H = 64;
  const int nN = in_sizes[0] / IN_CH;          // 100000
  const int nE = in_sizes[1] / 2;              // 1600000
  const int* src = ei;
  const int* dst = ei + nE;

  // workspace carve (256B aligned)
  char* ws = (char*)d_ws;
  size_t off = 0;
  auto carve = [&](size_t bytes) -> void* {
    void* p = ws + off;
    off += (bytes + 255) & ~(size_t)255;
    return p;
  };
  float*    dinv  = (float*)carve((size_t)nN * sizeof(float));      // deg -> dinv
  float*    hx1   = (float*)carve((size_t)nN * H * sizeof(float));  // x@Wg1; reused as decode hidden
  float*    acc1  = (float*)carve((size_t)nN * H * sizeof(float));  // conv1 accum -> h1
  float*    hx2   = (float*)carve((size_t)nN * Z * sizeof(float));  // h1@Wg2
  float*    acc2  = (float*)carve((size_t)nN * Z * sizeof(float));  // conv2 accum -> z
  _Float16* Wg1pk = (_Float16*)carve((size_t)(H/16)*(IN_CH/32)*512 * sizeof(_Float16));
  _Float16* Wg2pk = (_Float16*)carve((size_t)(Z/16)*(H/32)*512 * sizeof(_Float16));
  _Float16* Wf1pk = (_Float16*)carve((size_t)(H/16)*(Z/32)*512 * sizeof(_Float16));
  _Float16* Wf2pk = (_Float16*)carve((size_t)(IN_CH/16)*(H/32)*512 * sizeof(_Float16));

  auto cdiv = [](long long a, long long b) { return (int)((a + b - 1) / b); };
  dim3 blk(TPB);

  // init (re-zero every call: deterministic, no carried state)
  k_fill<<<cdiv(nN, TPB), blk, 0, stream>>>(dinv, nN, 1.0f);                 // self-loop degree
  k_fill<<<cdiv((long long)nN * H, TPB), blk, 0, stream>>>(acc1, (long long)nN * H, 0.0f);
  k_fill<<<cdiv((long long)nN * Z, TPB), blk, 0, stream>>>(acc2, (long long)nN * Z, 0.0f);

  // weight pre-pack into WMMA B-fragment layout
  k_prepack<<<cdiv((H/16)*(IN_CH/32)*512, TPB), blk, 0, stream>>>(Wg1, Wg1pk, IN_CH, H);
  k_prepack<<<cdiv((Z/16)*(H/32)*512,    TPB), blk, 0, stream>>>(Wg2, Wg2pk, H, Z);
  k_prepack<<<cdiv((H/16)*(Z/32)*512,    TPB), blk, 0, stream>>>(Wf1, Wf1pk, Z, H);
  k_prepack<<<cdiv((IN_CH/16)*(H/32)*512,TPB), blk, 0, stream>>>(Wf2, Wf2pk, H, IN_CH);

  // symmetric normalization: deg -> deg^{-1/2}
  k_degcount<<<cdiv(nE, TPB), blk, 0, stream>>>(dst, dinv, nE);
  k_rsqrt<<<cdiv(nN, TPB), blk, 0, stream>>>(dinv, nN);

  // conv1: hx1 = x @ Wg1 ; scatter ; h1 = relu(acc1 + self + bg1) (in place)
  k_gemm<4, 0><<<dim3(cdiv(nN, 128), 1), blk, 0, stream>>>(x, Wg1pk, hx1, nullptr, nN, IN_CH, H);
  k_scatter<<<cdiv((long long)nE * H, TPB), blk, 0, stream>>>(hx1, src, dst, dinv, acc1, nE, H, 6);
  k_finalize<<<cdiv((long long)nN * H, TPB), blk, 0, stream>>>(hx1, acc1, dinv, bg1, acc1, nN, H, 6, 1);

  // conv2: hx2 = h1 @ Wg2 ; scatter ; z = acc2 + self + bg2 (in place, no relu)
  k_gemm<2, 0><<<dim3(cdiv(nN, 128), 1), blk, 0, stream>>>(acc1, Wg2pk, hx2, nullptr, nN, H, Z);
  k_scatter<<<cdiv((long long)nE * Z, TPB), blk, 0, stream>>>(hx2, src, dst, dinv, acc2, nE, Z, 5);
  k_finalize<<<cdiv((long long)nN * Z, TPB), blk, 0, stream>>>(hx2, acc2, dinv, bg2, acc2, nN, Z, 5, 0);

  // decode: d = relu(z @ Wf1 + bf1) into hx1 ; out = relu(d @ Wf2 + bf2)
  k_gemm<4, 3><<<dim3(cdiv(nN, 128), 1), blk, 0, stream>>>(acc2, Wf1pk, hx1, bf1, nN, Z, H);
  k_gemm<4, 3><<<dim3(cdiv(nN, 128), IN_CH / 64), blk, 0, stream>>>(hx1, Wf2pk, (float*)d_out, bf2, nN, H, IN_CH);
}
